// SpatioTemporalLightningEncoder_34565896798848
// MI455X (gfx1250) — matrix operand order repaired and checked by reading, not compiled, not run on hardware
//
#include <hip/hip_runtime.h>

// ===========================================================================
// SpatioTemporalLightningEncoder forward for MI455X (gfx1250, wave32, WMMA).
// All heavy GEMMs run on v_wmma_f32_16x16x32_f16 (f16 in, f32 accumulate).
// Residual stream / LayerNorm stay fp32. Weights are repacked f16 N x K
// (B-matrix layout wants contiguous K per column lane).
// Workspace requirement: ~650 MB (see bump allocator below).
// ===========================================================================

typedef _Float16 v8h  __attribute__((ext_vector_type(8)));
typedef _Float16 v16h __attribute__((ext_vector_type(16)));
typedef float    v8f  __attribute__((ext_vector_type(8)));

// ---- problem constants ----
constexpr int B_    = 32;
constexpr int HIST_ = 10;
constexpr int Cc_   = 11;
constexpr int Hh_   = 15;
constexpr int Ww_   = 20;
constexpr int D_    = 256;
constexpr int NH_   = 4;
constexpr int HD_   = 64;
constexpr int NL_   = 4;
constexpr int ADIM_ = 20;
constexpr int OUT_  = 256;
constexpr int L_    = HIST_*Hh_*Ww_ + 1;     // 3001
constexpr int VIS_  = HIST_*Cc_*Hh_*Ww_;     // 33000
constexpr int BL_   = B_*L_;                 // 96032
constexpr int LP_   = 3008;                  // L padded to multiple of 32
constexpr int DFF_  = 4*D_;                  // 1024

__device__ inline v8f vz8() {
  v8f v;
#pragma unroll
  for (int i = 0; i < 8; ++i) v[i] = 0.0f;
  return v;
}
__device__ inline v16h cat16(v8h lo, v8h hi) {
  return __builtin_shufflevector(lo, hi, 0,1,2,3,4,5,6,7,8,9,10,11,12,13,14,15);
}

// ===========================================================================
// Generic batched GEMM:  D[M,N] = act(alpha*A[M,K]*Bt[N,K]^T + bias) + resid
// A row-major f16 (lda), Bt row-major f16 (N rows of K), f32 and/or f16 out.
// Block = 256 threads = 8 waves; block tile 128(M) x 64(N); wave tile 32x32.
// ===========================================================================
__global__ __launch_bounds__(256)
void gemm_wmma_f16(const _Float16* __restrict__ A, const _Float16* __restrict__ Bt,
                   const float* __restrict__ bias, const float* __restrict__ resid,
                   float* __restrict__ Cf, _Float16* __restrict__ Ch,
                   int M, int N, int K, int lda, int ldc, int ldch,
                   long long sAb, long long sAh, long long sBb, long long sBh,
                   long long sCfb, long long sChb, long long sChh,
                   int subBatch, float alpha, int act)
{
  const int z  = blockIdx.z;
  const int zb = z / subBatch, zh = z % subBatch;
  A  += zb*sAb + zh*sAh;
  Bt += zb*sBb + zh*sBh;
  if (Cf)    Cf    += zb*sCfb;
  if (resid) resid += zb*sCfb;
  if (Ch)    Ch    += zb*sChb + zh*sChh;

  const int wid    = threadIdx.x >> 5;
  const int lane   = threadIdx.x & 31;
  const int waveN  = wid & 1;
  const int waveM  = wid >> 1;
  const int rowBase = blockIdx.y*128 + waveM*32;
  const int colBase = blockIdx.x*64  + waveN*32;
  const int laneR  = lane & 15;
  const int laneHi = lane >> 4;

  v8f acc[2][2];
  acc[0][0] = vz8(); acc[0][1] = vz8(); acc[1][0] = vz8(); acc[1][1] = vz8();

  // A layout (16-bit A 16x32): lane holds row=laneR(+16*mi), two 16B chunks
  // at K offsets laneHi*8 and 16+laneHi*8.
  int r0 = rowBase + laneR;       if (r0 >= M) r0 = M - 1;
  int r1 = rowBase + 16 + laneR;  if (r1 >= M) r1 = M - 1;
  const _Float16* a0p = A + (long long)r0*lda + laneHi*8;
  const _Float16* a1p = A + (long long)r1*lda + laneHi*8;
  // B layout (16-bit B 32x16): lane holds col=laneR(+16*ni), two 16B chunks
  // at K offsets laneHi*8 and 16+laneHi*8.
  const _Float16* b0p = Bt + (long long)(colBase + laneR)*K      + laneHi*8;
  const _Float16* b1p = Bt + (long long)(colBase + 16 + laneR)*K + laneHi*8;

  for (int k0 = 0; k0 < K; k0 += 32) {
    v16h a0 = cat16(*(const v8h*)(a0p + k0), *(const v8h*)(a0p + k0 + 16));
    v16h a1 = cat16(*(const v8h*)(a1p + k0), *(const v8h*)(a1p + k0 + 16));
    v16h b0 = cat16(*(const v8h*)(b0p + k0), *(const v8h*)(b0p + k0 + 16));
    v16h b1 = cat16(*(const v8h*)(b1p + k0), *(const v8h*)(b1p + k0 + 16));
    if (k0 + 32 < K) {                       // gfx1250 global_prefetch_b8
      __builtin_prefetch(a0p + k0 + 32, 0, 1);
      __builtin_prefetch(b0p + k0 + 32, 0, 1);
    }
    acc[0][0] = __builtin_amdgcn_wmma_f32_16x16x32_f16(false, a0, false, b0, (short)0, acc[0][0], false, false);
    acc[0][1] = __builtin_amdgcn_wmma_f32_16x16x32_f16(false, a0, false, b1, (short)0, acc[0][1], false, false);
    acc[1][0] = __builtin_amdgcn_wmma_f32_16x16x32_f16(false, a1, false, b0, (short)0, acc[1][0], false, false);
    acc[1][1] = __builtin_amdgcn_wmma_f32_16x16x32_f16(false, a1, false, b1, (short)0, acc[1][1], false, false);
  }

  // C/D layout: lanes 0-15: VGPR r -> M=r ; lanes 16-31: VGPR r -> M=8+r.
#pragma unroll
  for (int mi = 0; mi < 2; ++mi) {
#pragma unroll
    for (int ni = 0; ni < 2; ++ni) {
      const int cN = colBase + ni*16 + laneR;
#pragma unroll
      for (int r = 0; r < 8; ++r) {
        const int rM = rowBase + mi*16 + laneHi*8 + r;
        if (rM < M) {
          float v = acc[mi][ni][r] * alpha;
          if (bias) v += bias[cN];
          if (act == 1) v = 0.5f * v * (1.0f + erff(v * 0.70710678118654752f)); // exact gelu
          if (resid) v += resid[(long long)rM*ldc + cN];
          if (Cf) Cf[(long long)rM*ldc + cN] = v;
          if (Ch) Ch[(long long)rM*ldch + cN] = (_Float16)v;
        }
      }
    }
  }
}

// ===========================================================================
// Wave-per-row LayerNorm over D=256 (wave32 shuffle reduction).
// ===========================================================================
__global__ __launch_bounds__(256)
void rowln_k(const float* __restrict__ in, long long rstride,
             const float* __restrict__ g, const float* __restrict__ bta,
             float* __restrict__ outF, _Float16* __restrict__ outH, int R)
{
  const int wid = threadIdx.x >> 5, lane = threadIdx.x & 31;
  const int row = blockIdx.x*8 + wid;
  if (row >= R) return;
  const float* p = in + (long long)row*rstride + lane*8;
  float v[8];
#pragma unroll
  for (int j = 0; j < 8; ++j) v[j] = p[j];
  float s = 0.f, q = 0.f;
#pragma unroll
  for (int j = 0; j < 8; ++j) { s += v[j]; q += v[j]*v[j]; }
#pragma unroll
  for (int off = 16; off >= 1; off >>= 1) {
    s += __shfl_xor(s, off, 32);
    q += __shfl_xor(q, off, 32);
  }
  const float mean = s * (1.0f/256.0f);
  const float var  = q * (1.0f/256.0f) - mean*mean;
  const float rstd = rsqrtf(var + 1e-5f);
#pragma unroll
  for (int j = 0; j < 8; ++j) {
    const int c = lane*8 + j;
    const float o = (v[j] - mean)*rstd*g[c] + bta[c];
    if (outF) outF[(long long)row*256 + c] = o;
    if (outH) outH[(long long)row*256 + c] = (_Float16)o;
  }
}

// ===========================================================================
// Patch projection (K=11, memory-bound) + cls token -> x fp32 (BL x 256).
// ===========================================================================
__global__ __launch_bounds__(256)
void patch_proj_k(const float* __restrict__ obs, const float* __restrict__ pw,
                  const float* __restrict__ pb, const float* __restrict__ cls,
                  float* __restrict__ x)
{
  const int idx = blockIdx.x*256 + threadIdx.x;      // BL_*256 total
  const int d  = idx & 255;
  const int bl = idx >> 8;
  if (bl >= BL_) return;
  const int b = bl / L_;
  const int l = bl % L_;
  float v;
  if (l == 0) {
    v = cls[d];
  } else {
    const int p    = l - 1;
    const int hist = p / (Hh_*Ww_);
    const int rem  = p % (Hh_*Ww_);
    const int hh   = rem / Ww_;
    const int ww   = rem % Ww_;
    const float* ob = obs + (long long)b*(VIS_+ADIM_);
    float s = pb[d];
#pragma unroll
    for (int c = 0; c < Cc_; ++c)
      s += ob[((hist*Cc_ + c)*Hh_ + hh)*Ww_ + ww] * pw[c*D_ + d];
    v = s;
  }
  x[idx] = v;
}

// ===========================================================================
// RoPE + SiLU + pack f16: q -> (b,h,l,d) row-major; k -> (b,h,d,l) transposed.
// Rows l in [L, LP) are zero-padded (safe for K-padded WMMA GEMMs).
// ===========================================================================
__global__ __launch_bounds__(256)
void rope_pack_k(const float* __restrict__ in, _Float16* __restrict__ out, int transposed)
{
  const int idx = blockIdx.x*256 + threadIdx.x;      // B*NH*LP*HD total
  const int e  = idx & 63;
  const int l  = (idx >> 6) % LP_;
  const int bh = idx / (64*LP_);
  if (bh >= B_*NH_) return;
  const long long oidx = transposed
      ? (long long)(bh*64 + e)*LP_ + l
      : ((long long)bh*LP_ + l)*64 + e;
  float v = 0.0f;
  if (l < L_) {
    const int b = bh >> 2, h = bh & 3;
    const float* row = in + ((long long)(b*L_ + l))*D_ + h*HD_;
    const float xv = row[e];
    const int j = e & 31;
    const float freq = __expf(-0.28782313662425572f * (float)j); // 10000^(-2j/64)
    const float ang  = (float)l * freq;
    const float c = cosf(ang), s = sinf(ang);
    const float other = (e < 32) ? row[e + 32] : row[e - 32];
    const float r = (e < 32) ? (xv*c - other*s) : (xv*c + other*s);
    v = r / (1.0f + __expf(-r));                     // silu
  }
  out[oidx] = (_Float16)v;
}

// v (f16, BL x 256) -> v_t f16 (b,h,d,l) transposed, zero-padded l >= L.
__global__ __launch_bounds__(256)
void pack_v_k(const _Float16* __restrict__ vh, _Float16* __restrict__ out)
{
  const int idx = blockIdx.x*256 + threadIdx.x;
  const int e  = idx & 63;
  const int l  = (idx >> 6) % LP_;
  const int bh = idx / (64*LP_);
  if (bh >= B_*NH_) return;
  const int b = bh >> 2, h = bh & 3;
  _Float16 v = (_Float16)0.0f;
  if (l < L_) v = vh[((long long)(b*L_ + l))*D_ + h*HD_ + e];
  out[(long long)(bh*64 + e)*LP_ + l] = v;
}

// fp32 weight (K x N) -> f16 transposed (N x K), per layer (blockIdx.z).
__global__ __launch_bounds__(256)
void packW_k(const float* __restrict__ W, _Float16* __restrict__ out, int K, int N)
{
  const int idx = blockIdx.x*256 + threadIdx.x;
  if (idx >= K*N) return;
  const int k = idx / N, n = idx % N;
  const long long base = (long long)blockIdx.z * K * N;
  out[base + (long long)n*K + k] = (_Float16)W[base + idx];
}

// Final head: out[b,o] = concat(cls_ln[b], action[b]) @ out_w + out_b (M=32).
__global__ __launch_bounds__(256)
void head_k(const float* __restrict__ clsln, const float* __restrict__ obs,
            const float* __restrict__ ow, const float* __restrict__ ob,
            float* __restrict__ out)
{
  const int idx = blockIdx.x*256 + threadIdx.x;      // B*OUT total
  if (idx >= B_*OUT_) return;
  const int o = idx & 255;
  const int b = idx >> 8;
  float s = ob[o];
  const float* c = clsln + b*D_;
  for (int j = 0; j < D_; ++j) s += c[j] * ow[j*OUT_ + o];
  const float* a = obs + (long long)b*(VIS_+ADIM_) + VIS_;
#pragma unroll
  for (int j = 0; j < ADIM_; ++j) s += a[j] * ow[(D_ + j)*OUT_ + o];
  out[idx] = s;
}

// ===========================================================================
// Host orchestration
// ===========================================================================
extern "C" void kernel_launch(void* const* d_in, const int* in_sizes, int n_in,
                              void* d_out, int out_size, void* d_ws, size_t ws_size,
                              hipStream_t stream)
{
  (void)in_sizes; (void)n_in; (void)out_size; (void)ws_size;

  const float* obs    = (const float*)d_in[0];
  const float* proj_w = (const float*)d_in[1];
  const float* proj_b = (const float*)d_in[2];
  const float* cls    = (const float*)d_in[3];
  const float* qw     = (const float*)d_in[4];
  const float* kw     = (const float*)d_in[5];
  const float* vw     = (const float*)d_in[6];
  const float* ow     = (const float*)d_in[7];
  const float* gq     = (const float*)d_in[8];
  const float* bq     = (const float*)d_in[9];
  const float* gk     = (const float*)d_in[10];
  const float* bk     = (const float*)d_in[11];
  const float* n1g    = (const float*)d_in[12];
  const float* n1b    = (const float*)d_in[13];
  const float* n2g    = (const float*)d_in[14];
  const float* n2b    = (const float*)d_in[15];
  const float* w1     = (const float*)d_in[16];
  const float* b1     = (const float*)d_in[17];
  const float* w2     = (const float*)d_in[18];
  const float* b2     = (const float*)d_in[19];
  const float* fng    = (const float*)d_in[20];
  const float* fnb    = (const float*)d_in[21];
  const float* out_w  = (const float*)d_in[22];
  const float* out_b  = (const float*)d_in[23];
  float* out = (float*)d_out;

  // ---- workspace bump allocator (256B aligned); total ~650 MB ----
  char* wsp = (char*)d_ws;
  auto alloc = [&](size_t bytes) -> void* {
    void* p = (void*)wsp;
    wsp += (bytes + 255) & ~(size_t)255;
    return p;
  };
  float*    x     = (float*)   alloc((size_t)BL_*D_*4);          // residual stream
  float*    t0    = (float*)   alloc((size_t)BL_*D_*4);          // q/k linear scratch
  _Float16* xln   = (_Float16*)alloc((size_t)BL_*D_*2);          // LN1/LN2 f16 out
  _Float16* vh    = (_Float16*)alloc((size_t)BL_*D_*2);          // v f16, reused as attnA
  _Float16* qp    = (_Float16*)alloc((size_t)B_*NH_*LP_*HD_*2);  // q packed (b,h,l,d)
  _Float16* kp    = (_Float16*)alloc((size_t)B_*NH_*LP_*HD_*2);  // k packed (b,h,d,l)
  _Float16* vp    = (_Float16*)alloc((size_t)B_*NH_*LP_*HD_*2);  // v packed (b,h,d,l)
  _Float16* hid   = (_Float16*)alloc((size_t)BL_*DFF_*2);        // FFN hidden
  _Float16* kvT   = (_Float16*)alloc((size_t)B_*NH_*HD_*HD_*2);  // kv^T per (b,h)
  float*    clsln = (float*)   alloc((size_t)B_*D_*4);
  _Float16* qwT   = (_Float16*)alloc((size_t)NL_*D_*D_*2);
  _Float16* kwT   = (_Float16*)alloc((size_t)NL_*D_*D_*2);
  _Float16* vwT   = (_Float16*)alloc((size_t)NL_*D_*D_*2);
  _Float16* owT   = (_Float16*)alloc((size_t)NL_*D_*D_*2);
  _Float16* w1T   = (_Float16*)alloc((size_t)NL_*D_*DFF_*2);
  _Float16* w2T   = (_Float16*)alloc((size_t)NL_*DFF_*D_*2);

  auto packW = [&](const float* W, _Float16* T, int K, int N) {
    dim3 g((K*N + 255)/256, 1, NL_);
    packW_k<<<g, 256, 0, stream>>>(W, T, K, N);
  };
  packW(qw, qwT, D_, D_);
  packW(kw, kwT, D_, D_);
  packW(vw, vwT, D_, D_);
  packW(ow, owT, D_, D_);
  packW(w1, w1T, D_, DFF_);
  packW(w2, w2T, DFF_, D_);

  // initial x = [cls ; vision @ proj_w + proj_b]
  patch_proj_k<<<dim3(BL_), 256, 0, stream>>>(obs, proj_w, proj_b, cls, x);

  auto gemm = [&](const _Float16* A, const _Float16* Bt, const float* bias,
                  const float* resid, float* Cf, _Float16* Ch,
                  int M, int N, int K, int lda, int ldc, int ldch,
                  long long sAb, long long sAh, long long sBb, long long sBh,
                  long long sCfb, long long sChb, long long sChh,
                  int batches, int subBatch, float alpha, int act) {
    dim3 g((N + 63)/64, (M + 127)/128, batches);
    gemm_wmma_f16<<<g, 256, 0, stream>>>(A, Bt, bias, resid, Cf, Ch,
                                         M, N, K, lda, ldc, ldch,
                                         sAb, sAh, sBb, sBh, sCfb, sChb, sChh,
                                         subBatch, alpha, act);
  };
  auto rowln = [&](const float* in, long long rstride, const float* g,
                   const float* b, float* oF, _Float16* oH, int R) {
    rowln_k<<<dim3((R + 7)/8), 256, 0, stream>>>(in, rstride, g, b, oF, oH, R);
  };

  const int packBlocks = (B_*NH_*LP_*HD_ + 255)/256;
  const long long sQb = (long long)NH_*LP_*HD_, sQh = (long long)LP_*HD_;
  const long long sKV = (long long)HD_*LP_;     // per-(b,h) stride of kp/vp

  for (int i = 0; i < NL_; ++i) {
    const long long wOff = (long long)i*D_*D_;
    // --- attention block ---
    rowln(x, D_, n1g + i*D_, n1b + i*D_, nullptr, xln, BL_);         // LN1 -> f16
    // q
    gemm(xln, qwT + wOff, nullptr, nullptr, t0, nullptr,
         BL_, D_, D_, D_, D_, 0, 0,0,0,0,0,0,0, 1, 1, 1.0f, 0);
    rowln(t0, D_, gq + i*D_, bq + i*D_, t0, nullptr, BL_);           // q LN in-place
    rope_pack_k<<<dim3(packBlocks), 256, 0, stream>>>(t0, qp, 0);
    // k
    gemm(xln, kwT + wOff, nullptr, nullptr, t0, nullptr,
         BL_, D_, D_, D_, D_, 0, 0,0,0,0,0,0,0, 1, 1, 1.0f, 0);
    rowln(t0, D_, gk + i*D_, bk + i*D_, t0, nullptr, BL_);
    rope_pack_k<<<dim3(packBlocks), 256, 0, stream>>>(t0, kp, 1);
    // v (straight to f16)
    gemm(xln, vwT + wOff, nullptr, nullptr, nullptr, vh,
         BL_, D_, D_, D_, D_, D_, 0,0,0,0,0,0,0, 1, 1, 1.0f, 0);
    pack_v_k<<<dim3(packBlocks), 256, 0, stream>>>(vh, vp);
    // kv^T[e,d] = sum_l v[l,e] k[l,d], scaled by 1/sqrt(HD)=0.125 (batched 128)
    gemm(vp, kp, nullptr, nullptr, nullptr, kvT,
         HD_, HD_, LP_, LP_, HD_, HD_,
         sKV, 0, sKV, 0, 0, (long long)HD_*HD_, 0,
         B_*NH_, 1, 0.125f, 0);
    // out[l,e] = q[l,:] @ kv[:,e]  -> attnA f16 (b,l, h*64+e), reuse vh
    gemm(qp, kvT, nullptr, nullptr, nullptr, vh,
         L_, HD_, HD_, HD_, HD_, D_,
         sQb, sQh, (long long)NH_*HD_*HD_, (long long)HD_*HD_,
         0, (long long)L_*D_, (long long)HD_,
         B_*NH_, NH_, 1.0f, 0);
    // o projection + residual into x
    gemm(vh, owT + wOff, nullptr, x, x, nullptr,
         BL_, D_, D_, D_, D_, 0, 0,0,0,0,0,0,0, 1, 1, 1.0f, 0);
    // --- FFN block ---
    rowln(x, D_, n2g + i*D_, n2b + i*D_, nullptr, xln, BL_);         // LN2 -> f16
    gemm(xln, w1T + (long long)i*D_*DFF_, b1 + i*DFF_, nullptr, nullptr, hid,
         BL_, DFF_, D_, D_, DFF_, DFF_, 0,0,0,0,0,0,0, 1, 1, 1.0f, 1); // gelu
    gemm(hid, w2T + (long long)i*DFF_*D_, b2 + i*D_, x, x, nullptr,
         BL_, D_, DFF_, DFF_, D_, 0, 0,0,0,0,0,0,0, 1, 1, 1.0f, 0);
  }

  // final LN on cls rows only (row stride L*D picks l==0 per batch)
  rowln(x, (long long)L_*D_, fng, fnb, clsln, nullptr, B_);
  head_k<<<dim3((B_*OUT_ + 255)/256), 256, 0, stream>>>(clsln, obs, out_w, out_b, out);
}
